// MLRelConcatClassifier_68384469287394
// MI455X (gfx1250) — matrix operand order repaired and checked by reading, not compiled, not run on hardware
//
#include <hip/hip_runtime.h>
#include <math.h>

#define B_   16
#define S_   256
#define H_   768
#define NPER 48
#define R_   40
#define E_   1536
#define NREL 19
#define N_   (B_*NPER)   /* 768 */

typedef __bf16 bf16;
typedef __attribute__((ext_vector_type(16))) __bf16 v16bf;
typedef __attribute__((ext_vector_type(8)))  float  v8f;
typedef __attribute__((ext_vector_type(4)))  unsigned int v4u;
typedef __attribute__((ext_vector_type(8)))  int v8i;
typedef __attribute__((ext_vector_type(4)))  int v4i;

// padded LDS row stride: 768 dwords + 3 pad dwords (TDM pad: 1 dword per 256)
#define AROW 771
#define ALDS 12336   /* 771*15 + 767 + 2  + 2 rounding margin */

// --------------------------------------------------------------- LDS offset
__device__ __forceinline__ unsigned lds_offset_u32(const void* p) {
  return (unsigned)(uintptr_t)(__attribute__((address_space(3))) void*)(p);
}

// ------------------------------------------------------ TDM descriptor pack
__device__ __forceinline__ v4u tdm_g0(unsigned ldsAddr, unsigned long long gaddr,
                                      bool gather) {
  v4u g;
  g[0] = 1u | (gather ? (1u << 31) : 0u);     // count=1; gather_mode; 16-bit idx
  g[1] = ldsAddr;                             // lds_addr (bytes)
  g[2] = (unsigned)(gaddr & 0xFFFFFFFFull);   // global_addr[31:0]
  g[3] = (unsigned)((gaddr >> 32) & 0x01FFFFFFull) | (2u << 30); // [56:32]|type=2
  return g;
}

__device__ __forceinline__ v8i tdm_g1(unsigned tdim0, unsigned tdim1,
                                      unsigned tile0, unsigned tile1,
                                      unsigned long long stride0) {
  // data_size=2 (4B); pad_enable=1; pad_interval code 7 (256 dw); pad_amount code 0 (1 dw)
  v8i g;
  g[0] = (int)((2u << 16) | (1u << 20) | (7u << 22));
  g[1] = (int)((tdim0 & 0xFFFFu) << 16);                       // tensor_dim0[15:0]
  g[2] = (int)((tdim0 >> 16) | ((tdim1 & 0xFFFFu) << 16));     // dim0 hi | dim1 lo
  g[3] = (int)((tdim1 >> 16) | ((tile0 & 0xFFFFu) << 16));     // dim1 hi | tile0
  g[4] = (int)(tile1 & 0xFFFFu);                               // tile1 | tile2=0
  g[5] = (int)(stride0 & 0xFFFFFFFFull);                       // stride0 lo
  g[6] = (int)((stride0 >> 32) & 0xFFFFull);                   // stride0 hi | stride1=0
  g[7] = 0;
  return g;
}

__device__ __forceinline__ void tdm_load(v4u g0, v8i g1, v4i g2, v4i g3) {
#if __clang_major__ >= 23
  v8i z = {0, 0, 0, 0, 0, 0, 0, 0};
  __builtin_amdgcn_tensor_load_to_lds(g0, g1, g2, g3, z, 0);
#else
  __builtin_amdgcn_tensor_load_to_lds(g0, g1, g2, g3, 0);
#endif
}

// ---------------------------------------------------------------- utilities
__global__ void zero_ints_kernel(int* __restrict__ p, int n) {
  int i = blockIdx.x * blockDim.x + threadIdx.x;
  if (i < n) p[i] = 0;
}

__global__ void count_kernel(const int* __restrict__ edst,
                             const int* __restrict__ etype,
                             int* __restrict__ cnt,      // [N*R]
                             int* __restrict__ relcnt) { // [R]
  int e = blockIdx.x * blockDim.x + threadIdx.x;
  if (e >= E_) return;
  atomicAdd(&cnt[(size_t)edst[e] * R_ + etype[e]], 1);
  atomicAdd(&relcnt[etype[e]], 1);
}

__global__ void scan_kernel(const int* __restrict__ relcnt,
                            int* __restrict__ relstart,  // [R+1]
                            int* __restrict__ reloff) {  // [R]
  if (threadIdx.x == 0 && blockIdx.x == 0) {
    int acc = 0;
    for (int r = 0; r < R_; ++r) {
      relstart[r] = acc;
      reloff[r]   = acc;
      acc += relcnt[r];
    }
    relstart[R_] = acc;
  }
}

__global__ void place_kernel(const int* __restrict__ etype,
                             int* __restrict__ reloff,
                             int* __restrict__ bucket) { // [E]
  int e = blockIdx.x * blockDim.x + threadIdx.x;
  if (e >= E_) return;
  int p = atomicAdd(&reloff[etype[e]], 1);
  bucket[p] = e;
}

// per-edge mean normalization 1/max(|N_r(dst)|,1), precomputed once
__global__ void norm_kernel(const int* __restrict__ edst,
                            const int* __restrict__ etype,
                            const int* __restrict__ cnt,
                            float* __restrict__ enorm) {
  int e = blockIdx.x * blockDim.x + threadIdx.x;
  if (e >= E_) return;
  int c = cnt[(size_t)edst[e] * R_ + etype[e]];
  enorm[e] = 1.0f / fmaxf((float)c, 1.0f);
}

// --------------------------------------------------- entity token mask-pool
__global__ void entity_kernel(const float* __restrict__ seq,
                              const float* __restrict__ m1,
                              const float* __restrict__ m2,
                              float* __restrict__ e1h,
                              float* __restrict__ e2h) {
  int idx = blockIdx.x * blockDim.x + threadIdx.x;
  if (idx >= B_ * H_) return;
  int b = idx / H_, h = idx % H_;
  float s1 = 0.f, s2 = 0.f;
  for (int s = 0; s < S_; ++s) {
    float v = seq[((size_t)b * S_ + s) * H_ + h];
    s1 += m1[b * S_ + s] * v;
    s2 += m2[b * S_ + s] * v;
  }
  e1h[idx] = s1;
  e2h[idx] = s2;
}

// --------------------------------------------------- ragged segment max-pool
__global__ void maxpool_kernel(const float* __restrict__ seq,
                               const float* __restrict__ depx,
                               const int* __restrict__ batch,
                               float* __restrict__ x0) {
  int idx = blockIdx.x * blockDim.x + threadIdx.x;
  if (idx >= N_ * H_) return;
  int n = idx / H_, h = idx % H_;
  int b = batch[n];
  float m = -INFINITY;
  for (int s = 0; s < S_; ++s)
    m = fmaxf(m, seq[((size_t)b * S_ + s) * H_ + h] + depx[(size_t)n * S_ + s]);
  x0[idx] = m;
}

// ------------------------------------------- WMMA GEMM: per-relation messages
// grid: (H/64, E/16, R)  block: 128 (4 waves).  A panel (16 gathered rows x 768)
// is DMA'd into LDS by the Tensor Data Mover in gather mode; K loop is sync-free.
__global__ __launch_bounds__(128)
void msg_gemm_kernel(const float* __restrict__ x,        // [N,H]
                     const float* __restrict__ Wrel,     // [R,H,H]
                     const int* __restrict__ bucket,     // [E] grouped edge ids
                     const int* __restrict__ relstart,   // [R+1]
                     const int* __restrict__ esrc,
                     const float* __restrict__ enorm,    // [E]
                     float* __restrict__ msg) {          // [E,H]
  const int r  = blockIdx.z;
  const int t0 = relstart[r];
  const int t1 = relstart[r + 1];
  const int m0 = blockIdx.y * 16;
  if (t0 + m0 >= t1) return;               // no edges in this tile
  const int n0   = blockIdx.x * 64;
  const int lane = threadIdx.x & 31;
  const int wave = threadIdx.x >> 5;

  __shared__ float As[ALDS];               // padded f32 A panel (TDM dest)
  __shared__ int   sEdge[16];
  __shared__ int   sRow[16];

  if (threadIdx.x < 16) {
    int mi = t0 + m0 + threadIdx.x;
    if (mi < t1) {
      int e = bucket[mi];
      sEdge[threadIdx.x] = e;
      sRow[threadIdx.x]  = esrc[e];
    } else {
      sEdge[threadIdx.x] = -1;
      sRow[threadIdx.x]  = esrc[bucket[t0]];  // safe dummy row (output dropped)
    }
  }
  __syncthreads();

  if (wave == 0) {
    // gather-mode TDM: 16x 16-bit row indices in groups 2/3
    unsigned id[16];
#pragma unroll
    for (int t = 0; t < 16; ++t) id[t] = (unsigned)sRow[t] & 0xFFFFu;
    v4i g2, g3;
#pragma unroll
    for (int q = 0; q < 4; ++q) {
      g2[q] = (int)(id[2 * q]     | (id[2 * q + 1] << 16));
      g3[q] = (int)(id[8 + 2 * q] | (id[9 + 2 * q] << 16));
    }
    v4u g0 = tdm_g0(lds_offset_u32(As), (unsigned long long)(uintptr_t)x, true);
    v8i g1 = tdm_g1(/*tdim0*/H_, /*tdim1*/N_, /*tile0*/H_, /*tile1(#idx)*/16,
                    /*stride0*/H_);
    tdm_load(g0, g1, g2, g3);
    __builtin_amdgcn_s_wait_tensorcnt(0);
  }
  __syncthreads();

  v8f c = {0.f, 0.f, 0.f, 0.f, 0.f, 0.f, 0.f, 0.f};
  const float* Wr = Wrel + (size_t)r * H_ * H_;
  const int kbase = (lane < 16) ? 0 : 8;
  const int M     = lane & 15;

  for (int k0 = 0; k0 < H_; k0 += 32) {
    v16bf a;
#pragma unroll
    for (int j = 0; j < 16; ++j) {
      int cc = k0 + kbase + j + ((j >= 8) ? 8 : 0);   // A 16x32 fragment layout
      a[j] = (bf16)As[AROW * M + cc + (cc >> 8)];     // padded LDS addressing
    }
    v16bf b;   // B fragment: lane = N column, elements = K rows in A's table order
    const float* wrow = Wr + (size_t)k0 * H_ + n0 + wave * 16 + (lane & 15);
#pragma unroll
    for (int j = 0; j < 16; ++j) b[j] = (bf16)wrow[(size_t)(kbase + j + ((j >= 8) ? 8 : 0)) * H_];

    c = __builtin_amdgcn_wmma_f32_16x16x32_bf16(false, a, false, b,
                                                (short)0, c, false, false);
  }

  const int col  = n0 + wave * 16 + (lane & 15);
  const int moff = (lane < 16) ? 0 : 8;
#pragma unroll
  for (int i = 0; i < 8; ++i) {
    int e = sEdge[i + moff];
    if (e >= 0) msg[(size_t)e * H_ + col] = c[i] * enorm[e];
  }
}

// ---------------------------------------------------- WMMA GEMM: root term
// grid: (H/64, N/16)  block: 128.  A panel (16 contiguous rows) via 2D TDM tile.
__global__ __launch_bounds__(128)
void root_gemm_kernel(const float* __restrict__ x,   // [N,H]
                      const float* __restrict__ W,   // [H,H]
                      float* __restrict__ out) {     // [N,H]
  const int m0   = blockIdx.y * 16;
  const int n0   = blockIdx.x * 64;
  const int lane = threadIdx.x & 31;
  const int wave = threadIdx.x >> 5;

  __shared__ float As[ALDS];

  if (wave == 0) {
    unsigned long long gaddr =
        (unsigned long long)(uintptr_t)x + (unsigned long long)m0 * H_ * 4ull;
    v4u g0 = tdm_g0(lds_offset_u32(As), gaddr, false);
    v8i g1 = tdm_g1(/*tdim0*/H_, /*tdim1*/16, /*tile0*/H_, /*tile1*/16,
                    /*stride0*/H_);
    v4i g2, g3;
    g2[0] = 1; g2[1] = 1; g2[2] = 0; g2[3] = 0;  // tensor_dim2=1, dim3=1 (unused)
    g3[0] = 0; g3[1] = 0; g3[2] = 0; g3[3] = 0;
    tdm_load(g0, g1, g2, g3);
    __builtin_amdgcn_s_wait_tensorcnt(0);
  }
  __syncthreads();

  v8f c = {0.f, 0.f, 0.f, 0.f, 0.f, 0.f, 0.f, 0.f};
  const int kbase = (lane < 16) ? 0 : 8;
  const int M     = lane & 15;

  for (int k0 = 0; k0 < H_; k0 += 32) {
    v16bf a;
#pragma unroll
    for (int j = 0; j < 16; ++j) {
      int cc = k0 + kbase + j + ((j >= 8) ? 8 : 0);
      a[j] = (bf16)As[AROW * M + cc + (cc >> 8)];
    }
    v16bf b;
    const float* wrow = W + (size_t)k0 * H_ + n0 + wave * 16 + (lane & 15);
#pragma unroll
    for (int j = 0; j < 16; ++j) b[j] = (bf16)wrow[(size_t)(kbase + j + ((j >= 8) ? 8 : 0)) * H_];

    c = __builtin_amdgcn_wmma_f32_16x16x32_bf16(false, a, false, b,
                                                (short)0, c, false, false);
  }

  const int col  = n0 + wave * 16 + (lane & 15);
  const int moff = (lane < 16) ? 0 : 8;
#pragma unroll
  for (int i = 0; i < 8; ++i)
    out[(size_t)(m0 + i + moff) * H_ + col] = c[i];
}

// ------------------------------- deterministic per-node aggregation + ReLU
__global__ __launch_bounds__(256)
void agg_kernel(const float* __restrict__ root,
                const float* __restrict__ bias,
                const float* __restrict__ msg,
                const int* __restrict__ edst,
                float* __restrict__ xout) {
  __shared__ int elist[96];
  __shared__ int ecnt;
  const int n = blockIdx.x;
  if (threadIdx.x == 0) {
    int cc = 0;
    for (int e = 0; e < E_; ++e)
      if (edst[e] == n && cc < 96) elist[cc++] = e;   // ascending e: deterministic
    ecnt = cc;
  }
  __syncthreads();
  const int cn = ecnt;
  for (int h = threadIdx.x; h < H_; h += 256) {
    float acc = root[(size_t)n * H_ + h] + bias[h];
    for (int j = 0; j < cn; ++j) acc += msg[(size_t)elist[j] * H_ + h];
    xout[(size_t)n * H_ + h] = fmaxf(acc, 0.f);
  }
}

// ------------------------------------------- masked segment-sum extraction
__global__ void dep_kernel(const float* __restrict__ x2,
                           const float* __restrict__ n1m,
                           const float* __restrict__ n2m,
                           float* __restrict__ e1d,
                           float* __restrict__ e2d) {
  int idx = blockIdx.x * blockDim.x + threadIdx.x;
  if (idx >= B_ * H_) return;
  int b = idx / H_, h = idx % H_;
  float d1 = 0.f, d2 = 0.f;
  for (int j = 0; j < NPER; ++j) {
    int n = b * NPER + j;                    // batch = repeat(arange(B), NPER)
    float xv = x2[(size_t)n * H_ + h];
    d1 += n1m[n] * xv;
    d2 += n2m[n] * xv;
  }
  e1d[idx] = d1;
  e2d[idx] = d2;
}

// ------------------------------------------------ tanh(concat) -> rel_output
__global__ void relout_kernel(const float* __restrict__ pooled,
                              const float* __restrict__ e1h,
                              const float* __restrict__ e2h,
                              const float* __restrict__ e1d,
                              const float* __restrict__ e2d,
                              float* __restrict__ out) {
  int idx = blockIdx.x * blockDim.x + threadIdx.x;
  if (idx >= B_ * 5 * H_) return;
  int b = idx / (5 * H_), j = idx % (5 * H_);
  float v;
  if      (j <     H_) v = pooled[(size_t)b * H_ + j];
  else if (j < 2 * H_) v = e1h[(size_t)b * H_ + (j -     H_)];
  else if (j < 3 * H_) v = e2h[(size_t)b * H_ + (j - 2 * H_)];
  else if (j < 4 * H_) v = e1d[(size_t)b * H_ + (j - 3 * H_)];
  else                 v = e2d[(size_t)b * H_ + (j - 4 * H_)];
  out[idx] = tanhf(v);
}

// --------------------------------------------------------- classifier head
__global__ void logits_kernel(const float* __restrict__ relout, // [B,5H]
                              const float* __restrict__ Wcls,   // [5H,NREL]
                              const float* __restrict__ bcls,
                              float* __restrict__ logits) {     // [B,NREL]
  int idx = blockIdx.x * blockDim.x + threadIdx.x;
  if (idx >= B_ * NREL) return;
  int b = idx / NREL, cl = idx % NREL;
  float acc = bcls[cl];
  for (int j = 0; j < 5 * H_; ++j)
    acc += relout[(size_t)b * 5 * H_ + j] * Wcls[(size_t)j * NREL + cl];
  logits[idx] = acc;
}

// ------------------------------------------------------- workspace layout
static constexpr size_t SZ_NH   = (size_t)N_ * H_ * sizeof(float);
static constexpr size_t SZ_BH   = (size_t)B_ * H_ * sizeof(float);
static constexpr size_t OFF_X0   = 0;
static constexpr size_t OFF_X1   = OFF_X0  + SZ_NH;
static constexpr size_t OFF_X2   = OFF_X1  + SZ_NH;
static constexpr size_t OFF_ROOT = OFF_X2  + SZ_NH;
static constexpr size_t OFF_MSG  = OFF_ROOT + SZ_NH;
static constexpr size_t OFF_E1H  = OFF_MSG + (size_t)E_ * H_ * sizeof(float);
static constexpr size_t OFF_E2H  = OFF_E1H + SZ_BH;
static constexpr size_t OFF_E1D  = OFF_E2H + SZ_BH;
static constexpr size_t OFF_E2D  = OFF_E1D + SZ_BH;
static constexpr size_t OFF_ENRM = OFF_E2D + SZ_BH;                 // float [E]
static constexpr size_t OFF_CNT  = OFF_ENRM + (size_t)E_ * 4;       // int [N*R]
static constexpr size_t OFF_RELC = OFF_CNT  + (size_t)N_ * R_ * 4;  // int [R]
static constexpr size_t OFF_RELS = OFF_RELC + (size_t)R_ * 4;       // int [R+1]
static constexpr size_t OFF_RELO = OFF_RELS + (size_t)(R_ + 1) * 4; // int [R]
static constexpr size_t OFF_BUCK = OFF_RELO + (size_t)R_ * 4;       // int [E]

extern "C" void kernel_launch(void* const* d_in, const int* in_sizes, int n_in,
                              void* d_out, int out_size, void* d_ws, size_t ws_size,
                              hipStream_t stream) {
  (void)in_sizes; (void)n_in; (void)out_size; (void)ws_size;

  const float* seq    = (const float*)d_in[0];
  const float* pooled = (const float*)d_in[1];
  const float* e1m    = (const float*)d_in[2];
  const float* e2m    = (const float*)d_in[3];
  const float* depx   = (const float*)d_in[4];
  const float* n1m    = (const float*)d_in[5];
  const float* n2m    = (const float*)d_in[6];
  const int*   batch  = (const int*)d_in[7];
  const int*   esrc   = (const int*)d_in[8];
  const int*   edst   = esrc + E_;
  const int*   etype  = (const int*)d_in[9];
  const float* Wrel1  = (const float*)d_in[10];
  const float* Wroot1 = (const float*)d_in[11];
  const float* b1     = (const float*)d_in[12];
  const float* Wrel2  = (const float*)d_in[13];
  const float* Wroot2 = (const float*)d_in[14];
  const float* b2     = (const float*)d_in[15];
  const float* Wcls   = (const float*)d_in[16];
  const float* bcls   = (const float*)d_in[17];

  char* ws = (char*)d_ws;
  float* x0    = (float*)(ws + OFF_X0);
  float* x1    = (float*)(ws + OFF_X1);
  float* x2    = (float*)(ws + OFF_X2);
  float* rootb = (float*)(ws + OFF_ROOT);
  float* msg   = (float*)(ws + OFF_MSG);
  float* e1h   = (float*)(ws + OFF_E1H);
  float* e2h   = (float*)(ws + OFF_E2H);
  float* e1d   = (float*)(ws + OFF_E1D);
  float* e2d   = (float*)(ws + OFF_E2D);
  float* enorm = (float*)(ws + OFF_ENRM);
  int*   cnt   = (int*)(ws + OFF_CNT);
  int*   relc  = (int*)(ws + OFF_RELC);
  int*   rels  = (int*)(ws + OFF_RELS);
  int*   relo  = (int*)(ws + OFF_RELO);
  int*   buck  = (int*)(ws + OFF_BUCK);

  float* out_rel    = (float*)d_out;                 // [B, 5H]
  float* out_logits = out_rel + (size_t)B_ * 5 * H_; // [B, NREL]

  // --- bucketize edges by relation + per-(dst,rel) counts + per-edge norms
  const int nz = N_ * R_ + R_;   // cnt and relcnt are contiguous
  zero_ints_kernel<<<(nz + 255) / 256, 256, 0, stream>>>(cnt, nz);
  count_kernel<<<(E_ + 255) / 256, 256, 0, stream>>>(edst, etype, cnt, relc);
  scan_kernel<<<1, 32, 0, stream>>>(relc, rels, relo);
  place_kernel<<<(E_ + 255) / 256, 256, 0, stream>>>(etype, relo, buck);
  norm_kernel<<<(E_ + 255) / 256, 256, 0, stream>>>(edst, etype, cnt, enorm);

  // --- entity pools and graph embeddings
  entity_kernel<<<(B_ * H_ + 255) / 256, 256, 0, stream>>>(seq, e1m, e2m, e1h, e2h);
  maxpool_kernel<<<(N_ * H_ + 255) / 256, 256, 0, stream>>>(seq, depx, batch, x0);

  const dim3 gemmBlk(128);
  const dim3 msgGrid(H_ / 64, E_ / 16, R_);   // worst-case m-tiles; empties exit
  const dim3 rootGrid(H_ / 64, N_ / 16);

  // --- layer 1
  msg_gemm_kernel<<<msgGrid, gemmBlk, 0, stream>>>(x0, Wrel1, buck, rels, esrc, enorm, msg);
  root_gemm_kernel<<<rootGrid, gemmBlk, 0, stream>>>(x0, Wroot1, rootb);
  agg_kernel<<<N_, 256, 0, stream>>>(rootb, b1, msg, edst, x1);

  // --- layer 2
  msg_gemm_kernel<<<msgGrid, gemmBlk, 0, stream>>>(x1, Wrel2, buck, rels, esrc, enorm, msg);
  root_gemm_kernel<<<rootGrid, gemmBlk, 0, stream>>>(x1, Wroot2, rootb);
  agg_kernel<<<N_, 256, 0, stream>>>(rootb, b2, msg, edst, x2);

  // --- head
  dep_kernel<<<(B_ * H_ + 255) / 256, 256, 0, stream>>>(x2, n1m, n2m, e1d, e2d);
  relout_kernel<<<(B_ * 5 * H_ + 255) / 256, 256, 0, stream>>>(pooled, e1h, e2h, e1d, e2d, out_rel);
  logits_kernel<<<(B_ * NREL + 255) / 256, 256, 0, stream>>>(out_rel, Wcls, bcls, out_logits);
}
